// MGMP_60627758350641
// MI455X (gfx1250) — matrix operand order repaired and checked by reading, not compiled, not run on hardware
//
#include <hip/hip_runtime.h>

#define DIM 128
#define CTXN 50
#define LDP 136   // padded f16 LDS row stride: 272B = 68 dwords -> 4-bank skew/row, conflict-free A-frags

typedef __attribute__((ext_vector_type(16))) _Float16 v16h;
typedef __attribute__((ext_vector_type(8)))  float    v8f;

// ---------------- WMMA helpers (CDNA5 gfx1250, wave32) ----------------

__device__ __forceinline__ v8f wmma_f16(v16h a, v16h b, v8f c) {
    // D = A(16x32 f16) * B(32x16 f16) + C(16x16 f32)
    return __builtin_amdgcn_wmma_f32_16x16x32_f16(false, a, false, b,
                                                  (short)0, c, false, false);
}

// A fragment (16x32 f16) from LDS, row-major, row stride ld (padded).
// ISA layout: lanes 0-15: row=lane, k = kb+{0..7,16..23}; lanes 16-31: k = kb+{8..15,24..31}.
__device__ __forceinline__ v16h load_a_lds(const _Float16* base, int ld, int kb, int lane) {
    int rr = lane & 15, hh = lane >> 4;
    const _Float16* p = base + rr * ld + kb + hh * 8;
    v16h a;
#pragma unroll
    for (int j = 0; j < 8; ++j) { a[j] = p[j]; a[8 + j] = p[16 + j]; }
    return a;
}

// B fragment (32x16 f16) from pre-packed weights where Wp[(n)*ld + k] = B[k][n] (k contiguous).
// ISA layout: lanes 0-15: col=lane, k = kb+{0..7,16..23}; lanes 16-31: k = kb+{8..15,24..31}.
__device__ __forceinline__ v16h load_b_h16(const _Float16* Wp, int ld, int n0, int kb, int lane) {
    int rr = lane & 15, hh = lane >> 4;
    const _Float16* p = Wp + (n0 + rr) * ld + kb + hh * 8;
    v16h b;
#pragma unroll
    for (int j = 0; j < 8; ++j) { b[j] = p[j]; b[8 + j] = p[16 + j]; }
    return b;
}

__device__ __forceinline__ float wred_sum(float v) {
#pragma unroll
    for (int off = 16; off > 0; off >>= 1) v += __shfl_xor(v, off, 32);
    return v;
}

// ---------------- prep: pack GEMM weights to f16 in fragment-friendly layouts ----------------

__global__ void mgmp_pack_attn_w(const float* __restrict__ wa_in_w,
                                 const float* __restrict__ wa_out_w,
                                 _Float16* __restrict__ wqP,  // [n*128+k] = Wq[n][k]
                                 _Float16* __restrict__ wkP,  // [(h*128+n)*32+kl] = Wk[h*32+kl][n]
                                 _Float16* __restrict__ wvP,  // [n*128+k] = Wv[n][k]
                                 _Float16* __restrict__ woP)  // [n*128+k] = Wo[n][k]
{
    int i = blockIdx.x * 256 + threadIdx.x;   // 16384
    if (i >= 16384) return;
    int n = i >> 7, k = i & 127;
    wqP[i] = (_Float16)wa_in_w[n * DIM + k];
    wvP[i] = (_Float16)wa_in_w[(256 + n) * DIM + k];
    woP[i] = (_Float16)wa_out_w[i];
    int h = i >> 12, rem = i & 4095, nn = rem >> 5, kl = rem & 31;   // transpose Wk
    wkP[i] = (_Float16)wa_in_w[(DIM + h * 32 + kl) * DIM + nn];
}

__global__ void mgmp_pack_conv_w(const float* __restrict__ w, _Float16* __restrict__ wp, int Kw) {
    int i = blockIdx.x * 256 + threadIdx.x;
    int Ktot = DIM * Kw;
    if (i >= DIM * Ktot) return;
    int n = i / Ktot;
    int k = i - n * Ktot;
    int kk = k >> 7, ci = k & 127;                 // k-major: k = kk*128 + ci
    wp[i] = (_Float16)w[n * Ktot + ci * Kw + kk];
}

// ---------------- Kernel 1: fine-grained word->context attention ----------------
// One block = 16 tokens, 256 threads (8 waves).  doc16[t][d] written f16.

__global__ __launch_bounds__(256)
void mgmp_fine_attn(const float* __restrict__ word_emb,
                    const float* __restrict__ news_emb,
                    const _Float16* __restrict__ wqP,
                    const _Float16* __restrict__ wkP,
                    const _Float16* __restrict__ wvP,
                    const _Float16* __restrict__ woP,
                    const float* __restrict__ wa_in_b,   // [384]
                    const float* __restrict__ wa_out_b,  // [128]
                    const int*   __restrict__ tokens,    // [8192]
                    const int*   __restrict__ word_ctx,  // [V*50]
                    _Float16*    __restrict__ doc16)     // [8192*128]
{
    __shared__ _Float16 sQ[16 * LDP];
    __shared__ _Float16 sQP[16 * LDP];
    __shared__ float    sU[4][16 * DIM];
    __shared__ float    sConst[16][4];
    __shared__ _Float16 sW[4][16 * LDP];
    __shared__ _Float16 sVout[16 * LDP];

    const int tid  = threadIdx.x;
    const int wv   = tid >> 5;
    const int lane = tid & 31;
    const int rr   = lane & 15;
    const int hh   = lane >> 4;
    const int t0   = blockIdx.x * 16;
    const float scale = 0.17677669529663687f; // 1/sqrt(32)

    // phase 0: gather word embeddings (Q rows) -> f16 LDS
    for (int i = tid; i < 16 * DIM; i += 256) {
        int row = i >> 7, col = i & 127;
        sQ[row * LDP + col] = (_Float16)word_emb[tokens[t0 + row] * DIM + col];
    }
    __syncthreads();

    // phase 1: QP = Q @ Wq^T + bq
    {
        int n0 = wv * 16;
        v8f acc = {};
        float bb = wa_in_b[n0 + rr];
#pragma unroll
        for (int i = 0; i < 8; ++i) acc[i] = bb;
        for (int kb = 0; kb < DIM; kb += 32) {
            v16h a = load_a_lds(sQ, LDP, kb, lane);
            v16h b = load_b_h16(wqP, DIM, n0, kb, lane);
            acc = wmma_f16(a, b, acc);
        }
#pragma unroll
        for (int i = 0; i < 8; ++i)
            sQP[(hh * 8 + i) * LDP + n0 + rr] = (_Float16)acc[i];
    }
    __syncthreads();

    // phase 2: U_h = (QP_h) @ Wk_h, scaled by 1/sqrt(dh). 32 jobs over 8 waves, K=32 single step.
    for (int it = 0; it < 4; ++it) {
        int job = wv + 8 * it;
        int h = job >> 3;
        int n0 = (job & 7) * 16;
        v16h a = load_a_lds(sQP, LDP, h * 32, lane);
        v16h b = load_b_h16(wkP + h * 4096, 32, n0, 0, lane);  // pre-transposed, contiguous
        v8f acc = {};
        acc = wmma_f16(a, b, acc);
#pragma unroll
        for (int i = 0; i < 8; ++i)
            sU[h][(hh * 8 + i) * DIM + n0 + rr] = acc[i] * scale;
    }
    // per-token per-head score constant: (qh . bk) * scale
    if (tid < 64) {
        int t = tid >> 2, h = tid & 3;
        float c = 0.f;
        for (int k = 0; k < 32; ++k)
            c += (float)sQP[t * LDP + h * 32 + k] * wa_in_b[DIM + h * 32 + k];
        sConst[t][h] = c * scale;
    }
    __syncthreads();

    // phase 3: streaming online-softmax over 50 gathered ctx rows (2 tokens per wave)
    for (int rep = 0; rep < 2; ++rep) {
        int t = wv + rep * 8;
        int tok = tokens[t0 + t];
        const int* ctx = word_ctx + tok * CTXN;
        float4 u0 = *(const float4*)&sU[0][t * DIM + lane * 4];
        float4 u1 = *(const float4*)&sU[1][t * DIM + lane * 4];
        float4 u2 = *(const float4*)&sU[2][t * DIM + lane * 4];
        float4 u3 = *(const float4*)&sU[3][t * DIM + lane * 4];
        float c0 = sConst[t][0], c1 = sConst[t][1], c2 = sConst[t][2], c3 = sConst[t][3];
        float m0 = -1e30f, m1 = -1e30f, m2 = -1e30f, m3 = -1e30f;
        float d0 = 0.f, d1 = 0.f, d2 = 0.f, d3 = 0.f;
        float4 a0 = {0,0,0,0}, a1 = {0,0,0,0}, a2 = {0,0,0,0}, a3 = {0,0,0,0};
        for (int c = 0; c < CTXN; ++c) {
            int cid = ctx[c];
            if (c + 1 < CTXN)
                __builtin_prefetch(news_emb + ctx[c + 1] * DIM + lane * 4, 0, 1);
            float4 x = *(const float4*)(news_emb + cid * DIM + lane * 4);
            float s0 = u0.x*x.x + u0.y*x.y + u0.z*x.z + u0.w*x.w;
            float s1 = u1.x*x.x + u1.y*x.y + u1.z*x.z + u1.w*x.w;
            float s2 = u2.x*x.x + u2.y*x.y + u2.z*x.z + u2.w*x.w;
            float s3 = u3.x*x.x + u3.y*x.y + u3.z*x.z + u3.w*x.w;
            s0 = wred_sum(s0) + c0;  s1 = wred_sum(s1) + c1;
            s2 = wred_sum(s2) + c2;  s3 = wred_sum(s3) + c3;
            {   float mn = fmaxf(m0, s0); float cr = __expf(m0 - mn); float p = __expf(s0 - mn);
                d0 = d0*cr + p; a0.x = a0.x*cr + p*x.x; a0.y = a0.y*cr + p*x.y;
                a0.z = a0.z*cr + p*x.z; a0.w = a0.w*cr + p*x.w; m0 = mn; }
            {   float mn = fmaxf(m1, s1); float cr = __expf(m1 - mn); float p = __expf(s1 - mn);
                d1 = d1*cr + p; a1.x = a1.x*cr + p*x.x; a1.y = a1.y*cr + p*x.y;
                a1.z = a1.z*cr + p*x.z; a1.w = a1.w*cr + p*x.w; m1 = mn; }
            {   float mn = fmaxf(m2, s2); float cr = __expf(m2 - mn); float p = __expf(s2 - mn);
                d2 = d2*cr + p; a2.x = a2.x*cr + p*x.x; a2.y = a2.y*cr + p*x.y;
                a2.z = a2.z*cr + p*x.z; a2.w = a2.w*cr + p*x.w; m2 = mn; }
            {   float mn = fmaxf(m3, s3); float cr = __expf(m3 - mn); float p = __expf(s3 - mn);
                d3 = d3*cr + p; a3.x = a3.x*cr + p*x.x; a3.y = a3.y*cr + p*x.y;
                a3.z = a3.z*cr + p*x.z; a3.w = a3.w*cr + p*x.w; m3 = mn; }
        }
        float i0 = 1.f/d0, i1 = 1.f/d1, i2 = 1.f/d2, i3 = 1.f/d3;
        _Float16* w0 = &sW[0][t * LDP + lane * 4];
        _Float16* w1 = &sW[1][t * LDP + lane * 4];
        _Float16* w2 = &sW[2][t * LDP + lane * 4];
        _Float16* w3 = &sW[3][t * LDP + lane * 4];
        w0[0]=(_Float16)(a0.x*i0); w0[1]=(_Float16)(a0.y*i0); w0[2]=(_Float16)(a0.z*i0); w0[3]=(_Float16)(a0.w*i0);
        w1[0]=(_Float16)(a1.x*i1); w1[1]=(_Float16)(a1.y*i1); w1[2]=(_Float16)(a1.z*i1); w1[3]=(_Float16)(a1.w*i1);
        w2[0]=(_Float16)(a2.x*i2); w2[1]=(_Float16)(a2.y*i2); w2[2]=(_Float16)(a2.z*i2); w2[3]=(_Float16)(a2.w*i2);
        w3[0]=(_Float16)(a3.x*i3); w3[1]=(_Float16)(a3.y*i3); w3[2]=(_Float16)(a3.z*i3); w3[3]=(_Float16)(a3.w*i3);
    }
    __syncthreads();

    // phase 4: Vout_h = W_h @ Wv_h^T + bv_h   (8 jobs = 4 heads x 2 n-tiles)
    {
        int h = wv >> 1;
        int n0 = (wv & 1) * 16;
        v8f acc = {};
        float bb = wa_in_b[256 + h * 32 + n0 + rr];
#pragma unroll
        for (int i = 0; i < 8; ++i) acc[i] = bb;
        for (int kb = 0; kb < DIM; kb += 32) {
            v16h a = load_a_lds(sW[h], LDP, kb, lane);
            v16h b = load_b_h16(wvP + (h * 32) * DIM, DIM, n0, kb, lane);
            acc = wmma_f16(a, b, acc);
        }
#pragma unroll
        for (int i = 0; i < 8; ++i)
            sVout[(hh * 8 + i) * LDP + h * 32 + n0 + rr] = (_Float16)acc[i];
    }
    __syncthreads();

    // phase 5: doc = Vout @ Wo^T + bo  (stored f16 for the conv stage)
    {
        int n0 = wv * 16;
        v8f acc = {};
        float bb = wa_out_b[n0 + rr];
#pragma unroll
        for (int i = 0; i < 8; ++i) acc[i] = bb;
        for (int kb = 0; kb < DIM; kb += 32) {
            v16h a = load_a_lds(sVout, LDP, kb, lane);
            v16h b = load_b_h16(woP, DIM, n0, kb, lane);
            acc = wmma_f16(a, b, acc);
        }
#pragma unroll
        for (int i = 0; i < 8; ++i)
            doc16[(t0 + hh * 8 + i) * DIM + n0 + rr] = (_Float16)acc[i];
    }
}

// ---------------- Kernel 2: TextCNN conv (implicit GEMM) + ReLU + maxpool ----------------

__global__ __launch_bounds__(256)
void mgmp_conv_pool(const _Float16* __restrict__ doc16,
                    const _Float16* __restrict__ w3P, const float* __restrict__ b3,
                    const _Float16* __restrict__ w4P, const float* __restrict__ b4,
                    const _Float16* __restrict__ w5P, const float* __restrict__ b5,
                    float* __restrict__ pooled /* [32][3][128], f32 bits via atomicMax */)
{
    int lt  = blockIdx.x;            // 16 l-tiles
    int cid = blockIdx.y;            // conv 0..2 -> K=3,4,5
    int bb  = blockIdx.z;            // batch
    int Kw  = 3 + cid;
    const _Float16* W  = (cid == 0) ? w3P : (cid == 1) ? w4P : w5P;
    const float* bias  = (cid == 0) ? b3  : (cid == 1) ? b4  : b5;
    int Lout = 256 - Kw + 1;
    int tid = threadIdx.x, wv = tid >> 5, lane = tid & 31;
    int rr = lane & 15, hh = lane >> 4;
    int l0 = lt * 16, n0 = wv * 16;
    int Ktot = DIM * Kw;

    v8f acc = {};
    float bv = bias[n0 + rr];
#pragma unroll
    for (int i = 0; i < 8; ++i) acc[i] = bv;

    for (int kb = 0; kb < Ktot; kb += 32) {
        int kk = kb >> 7;            // kernel tap (constant within a 32-wide window)
        int cibase = kb & 127;
        // A: sliding-window rows of doc16 (contiguous f16)
        v16h a;
        int l = l0 + rr + kk;
        bool ok = (l < 256);
        const _Float16* arow = doc16 + (bb * 256 + (ok ? l : 0)) * DIM + cibase + hh * 8;
#pragma unroll
        for (int j = 0; j < 8; ++j) {
            a[j]     = ok ? arow[j]      : (_Float16)0.f;
            a[8 + j] = ok ? arow[16 + j] : (_Float16)0.f;
        }
        // B: pre-packed k-major weights, contiguous
        v16h bf;
        {
            const _Float16* p = W + (n0 + rr) * Ktot + kb + hh * 8;
#pragma unroll
            for (int j = 0; j < 8; ++j) { bf[j] = p[j]; bf[8 + j] = p[16 + j]; }
        }
        acc = wmma_f16(a, bf, acc);
    }
    // fused ReLU + tile max (relu >= 0 so out-of-range rows contribute harmless 0)
    float vmax = 0.f;
#pragma unroll
    for (int i = 0; i < 8; ++i) {
        int l = l0 + hh * 8 + i;
        float v = (l < Lout) ? fmaxf(acc[i], 0.f) : 0.f;
        vmax = fmaxf(vmax, v);
    }
    vmax = fmaxf(vmax, __shfl_xor(vmax, 16, 32));
    if (hh == 0) {
        unsigned* dst = (unsigned*)pooled + (bb * 3 + cid) * DIM + n0 + rr;
        atomicMax(dst, __float_as_uint(vmax));
    }
}

__global__ void mgmp_zero(float* p, int n) {
    int i = blockIdx.x * blockDim.x + threadIdx.x;
    if (i < n) p[i] = 0.f;
}

__global__ void mgmp_fc_semantic(const float* __restrict__ pooled,
                                 const float* __restrict__ fcw,
                                 const float* __restrict__ fcb,
                                 float* __restrict__ sem)
{
    int idx = blockIdx.x * blockDim.x + threadIdx.x;   // 32*128
    int b = idx >> 7, j = idx & 127;
    const unsigned* pb = (const unsigned*)pooled + b * 384;
    float a = fcb[j];
    for (int i = 0; i < 384; ++i) a += __uint_as_float(pb[i]) * fcw[j * 384 + i];
    sem[idx] = a;
}

// ---------------- Kernel 3a: GRU over 3-hop meta-paths ----------------

__global__ void mgmp_gru(const float* __restrict__ news_emb,
                         const float* __restrict__ mid_emb,
                         const int* __restrict__ news_ids,
                         const int* __restrict__ mid_idx,
                         const int* __restrict__ end_idx,
                         const float* __restrict__ wih, const float* __restrict__ whh,
                         const float* __restrict__ bih, const float* __restrict__ bhh,
                         float* __restrict__ enc /* [320][128] */)
{
    int bs = blockIdx.x;               // b*10+s
    int b  = bs / 10;
    int j  = threadIdx.x;              // 128 threads
    __shared__ float x[DIM], hA[DIM], hB[DIM];
    float* hc = hA; float* hn2 = hB;
    hA[j] = 0.f;
    __syncthreads();
    for (int t = 0; t < 3; ++t) {
        const float* src = (t == 0) ? (news_emb + news_ids[b] * DIM)
                         : (t == 1) ? (mid_emb + mid_idx[bs] * DIM)
                                    : (news_emb + end_idx[bs] * DIM);
        x[j] = src[j];
        __syncthreads();
        float gr = bih[j], gz = bih[DIM + j], gn = bih[2 * DIM + j];
        float hr = bhh[j], hz = bhh[DIM + j], hn = bhh[2 * DIM + j];
        for (int k = 0; k < DIM; ++k) {
            float xv = x[k], hv = hc[k];
            gr += xv * wih[j * DIM + k];
            gz += xv * wih[(DIM + j) * DIM + k];
            gn += xv * wih[(2 * DIM + j) * DIM + k];
            hr += hv * whh[j * DIM + k];
            hz += hv * whh[(DIM + j) * DIM + k];
            hn += hv * whh[(2 * DIM + j) * DIM + k];
        }
        float r = 1.f / (1.f + __expf(-(gr + hr)));
        float z = 1.f / (1.f + __expf(-(gz + hz)));
        float n = tanhf(gn + r * hn);
        hn2[j] = (1.f - z) * n + z * hc[j];
        __syncthreads();
        float* tmp = hc; hc = hn2; hn2 = tmp;
    }
    enc[bs * DIM + j] = hc[j];
}

// ---------------- Kernel 3b: MHSA over 10 path samples + residual + LN + mean ----------------

__global__ void mgmp_samp_attn(const float* __restrict__ enc,   // [32][10][128]
                               const float* __restrict__ aw, const float* __restrict__ ab,
                               const float* __restrict__ ow, const float* __restrict__ ob,
                               const float* __restrict__ lng, const float* __restrict__ lnb,
                               float* __restrict__ eout)        // [32][128]
{
    int b = blockIdx.x, j = threadIdx.x;       // 128 threads
    __shared__ float E[10][DIM], Q[10][DIM], K[10][DIM], V[10][DIM];
    __shared__ float A[4][10][10], O[10][DIM], R[10][DIM];
    __shared__ float mu[10], rstd[10];
    for (int s = 0; s < 10; ++s) E[s][j] = enc[(b * 10 + s) * DIM + j];
    __syncthreads();
    for (int s = 0; s < 10; ++s) {
        float q = ab[j], k = ab[DIM + j], v = ab[2 * DIM + j];
        for (int c = 0; c < DIM; ++c) {
            float e = E[s][c];
            q += e * aw[j * DIM + c];
            k += e * aw[(DIM + j) * DIM + c];
            v += e * aw[(2 * DIM + j) * DIM + c];
        }
        Q[s][j] = q; K[s][j] = k; V[s][j] = v;
    }
    __syncthreads();
    for (int idx = j; idx < 400; idx += 128) {
        int h = idx / 100, rem = idx % 100, qi = rem / 10, ki = rem % 10;
        float s = 0.f;
        for (int d = 0; d < 32; ++d) s += Q[qi][h * 32 + d] * K[ki][h * 32 + d];
        A[h][qi][ki] = s * 0.17677669529663687f;
    }
    __syncthreads();
    if (j < 40) {
        int h = j / 10, qi = j % 10;
        float mx = -1e30f;
        for (int k = 0; k < 10; ++k) mx = fmaxf(mx, A[h][qi][k]);
        float sm = 0.f;
        for (int k = 0; k < 10; ++k) { float e = __expf(A[h][qi][k] - mx); A[h][qi][k] = e; sm += e; }
        float inv = 1.f / sm;
        for (int k = 0; k < 10; ++k) A[h][qi][k] *= inv;
    }
    __syncthreads();
    {
        int h = j >> 5, d = j & 31;
        for (int qi = 0; qi < 10; ++qi) {
            float o = 0.f;
            for (int k = 0; k < 10; ++k) o += A[h][qi][k] * V[k][h * 32 + d];
            O[qi][j] = o;
        }
    }
    __syncthreads();
    for (int qi = 0; qi < 10; ++qi) {
        float o = ob[j];
        for (int c = 0; c < DIM; ++c) o += O[qi][c] * ow[j * DIM + c];
        R[qi][j] = E[qi][j] + o;
    }
    __syncthreads();
    if (j < 10) {
        float m = 0.f;
        for (int c = 0; c < DIM; ++c) m += R[j][c];
        m /= 128.f;
        float v2 = 0.f;
        for (int c = 0; c < DIM; ++c) { float d = R[j][c] - m; v2 += d * d; }
        mu[j] = m; rstd[j] = rsqrtf(v2 / 128.f + 1e-5f);
    }
    __syncthreads();
    float acc = 0.f;
    for (int qi = 0; qi < 10; ++qi)
        acc += (R[qi][j] - mu[qi]) * rstd[qi] * lng[j] + lnb[j];
    eout[b * DIM + j] = acc * 0.1f;
}

// ---------------- Kernel 4: meta-path attention + classifier ----------------

__global__ void mgmp_final(const float* __restrict__ sem,
                           const float* __restrict__ e0, const float* __restrict__ e1,
                           const float* __restrict__ mpw1, const float* __restrict__ mpb1,
                           const float* __restrict__ mpw2, const float* __restrict__ mpb2,
                           const float* __restrict__ cw1, const float* __restrict__ cb1,
                           const float* __restrict__ cw2, const float* __restrict__ cb2,
                           float* __restrict__ out)
{
    int b = blockIdx.x, j = threadIdx.x;   // 128 threads
    __shared__ float st[2][DIM], tt[2][64], wgt[2], comb[2 * DIM], hh[DIM];
    st[0][j] = e0[b * DIM + j];
    st[1][j] = e1[b * DIM + j];
    __syncthreads();
    {
        int m = j >> 6, u = j & 63;
        float t = mpb1[u];
        for (int k = 0; k < DIM; ++k) t += st[m][k] * mpw1[u * DIM + k];
        tt[m][u] = tanhf(t);
    }
    __syncthreads();
    if (j < 2) {
        float sc = mpb2[0];
        for (int u = 0; u < 64; ++u) sc += tt[j][u] * mpw2[u];
        wgt[j] = sc;
    }
    __syncthreads();
    if (j == 0) {
        float mx = fmaxf(wgt[0], wgt[1]);
        float ea = __expf(wgt[0] - mx), eb = __expf(wgt[1] - mx);
        float inv = 1.f / (ea + eb);
        wgt[0] = ea * inv; wgt[1] = eb * inv;
    }
    __syncthreads();
    comb[j] = sem[b * DIM + j];
    comb[DIM + j] = st[0][j] * wgt[0] + st[1][j] * wgt[1];
    __syncthreads();
    {
        float a = cb1[j];
        for (int k = 0; k < 2 * DIM; ++k) a += comb[k] * cw1[j * 2 * DIM + k];
        hh[j] = fmaxf(a, 0.f);
    }
    __syncthreads();
    if (j < 2) {
        float a = cb2[j];
        for (int k = 0; k < DIM; ++k) a += hh[k] * cw2[j * DIM + k];
        out[b * 2 + j] = a;
    }
}

// ---------------- launcher ----------------

extern "C" void kernel_launch(void* const* d_in, const int* in_sizes, int n_in,
                              void* d_out, int out_size, void* d_ws, size_t ws_size,
                              hipStream_t stream) {
    const float* news_emb   = (const float*)d_in[0];
    const float* user_emb   = (const float*)d_in[1];
    const float* source_emb = (const float*)d_in[2];
    const float* word_emb   = (const float*)d_in[3];
    const float* wa_in_w    = (const float*)d_in[4];
    const float* wa_in_b    = (const float*)d_in[5];
    const float* wa_out_w   = (const float*)d_in[6];
    const float* wa_out_b   = (const float*)d_in[7];
    const float* conv_w3    = (const float*)d_in[8];
    const float* conv_b3    = (const float*)d_in[9];
    const float* conv_w4    = (const float*)d_in[10];
    const float* conv_b4    = (const float*)d_in[11];
    const float* conv_w5    = (const float*)d_in[12];
    const float* conv_b5    = (const float*)d_in[13];
    const float* fc_w       = (const float*)d_in[14];
    const float* fc_b       = (const float*)d_in[15];
    const float* nsn_gru_wih = (const float*)d_in[16];
    const float* nsn_gru_whh = (const float*)d_in[17];
    const float* nsn_gru_bih = (const float*)d_in[18];
    const float* nsn_gru_bhh = (const float*)d_in[19];
    const float* nsn_aw = (const float*)d_in[20];
    const float* nsn_ab = (const float*)d_in[21];
    const float* nsn_ow = (const float*)d_in[22];
    const float* nsn_ob = (const float*)d_in[23];
    const float* nsn_lg = (const float*)d_in[24];
    const float* nsn_lb = (const float*)d_in[25];
    const float* nun_gru_wih = (const float*)d_in[26];
    const float* nun_gru_whh = (const float*)d_in[27];
    const float* nun_gru_bih = (const float*)d_in[28];
    const float* nun_gru_bhh = (const float*)d_in[29];
    const float* nun_aw = (const float*)d_in[30];
    const float* nun_ab = (const float*)d_in[31];
    const float* nun_ow = (const float*)d_in[32];
    const float* nun_ob = (const float*)d_in[33];
    const float* nun_lg = (const float*)d_in[34];
    const float* nun_lb = (const float*)d_in[35];
    const float* mp_w1 = (const float*)d_in[36];
    const float* mp_b1 = (const float*)d_in[37];
    const float* mp_w2 = (const float*)d_in[38];
    const float* mp_b2 = (const float*)d_in[39];
    const float* cls_w1 = (const float*)d_in[40];
    const float* cls_b1 = (const float*)d_in[41];
    const float* cls_w2 = (const float*)d_in[42];
    const float* cls_b2 = (const float*)d_in[43];
    const int* news_ids = (const int*)d_in[44];
    const int* news_word_indices = (const int*)d_in[45];
    const int* word_ctx_ids = (const int*)d_in[46];
    const int* nsn_mid = (const int*)d_in[47];
    const int* nsn_end = (const int*)d_in[48];
    const int* nun_mid = (const int*)d_in[49];
    const int* nun_end = (const int*)d_in[50];

    // workspace carve-up
    char* base = (char*)d_ws;
    _Float16* doc16 = (_Float16*)base;               base += 8192 * 128 * sizeof(_Float16);
    _Float16* wqP   = (_Float16*)base;               base += 16384 * sizeof(_Float16);
    _Float16* wkP   = (_Float16*)base;               base += 16384 * sizeof(_Float16);
    _Float16* wvP   = (_Float16*)base;               base += 16384 * sizeof(_Float16);
    _Float16* woP   = (_Float16*)base;               base += 16384 * sizeof(_Float16);
    _Float16* cw3P  = (_Float16*)base;               base += 128 * 384 * sizeof(_Float16);
    _Float16* cw4P  = (_Float16*)base;               base += 128 * 512 * sizeof(_Float16);
    _Float16* cw5P  = (_Float16*)base;               base += 128 * 640 * sizeof(_Float16);
    float* pooled   = (float*)base;                  base += 32 * 384 * sizeof(float);
    float* semantic = (float*)base;                  base += 32 * 128 * sizeof(float);
    float* enc      = (float*)base;                  base += 2 * 320 * 128 * sizeof(float);
    float* evec     = (float*)base;                  base += 2 * 32 * 128 * sizeof(float);

    // 0) pack weights to f16 fragment layouts
    mgmp_pack_attn_w<<<64, 256, 0, stream>>>(wa_in_w, wa_out_w, wqP, wkP, wvP, woP);
    mgmp_pack_conv_w<<<192, 256, 0, stream>>>(conv_w3, cw3P, 3);
    mgmp_pack_conv_w<<<256, 256, 0, stream>>>(conv_w4, cw4P, 4);
    mgmp_pack_conv_w<<<320, 256, 0, stream>>>(conv_w5, cw5P, 5);

    // 1) fine-grained word-context attention -> doc16 [8192][128]
    mgmp_fine_attn<<<512, 256, 0, stream>>>(word_emb, news_emb, wqP, wkP, wvP, woP,
                                            wa_in_b, wa_out_b, news_word_indices,
                                            word_ctx_ids, doc16);
    // 2) TextCNN
    mgmp_zero<<<(32 * 384 + 255) / 256, 256, 0, stream>>>(pooled, 32 * 384);
    dim3 cg(16, 3, 32);
    mgmp_conv_pool<<<cg, 256, 0, stream>>>(doc16, cw3P, conv_b3, cw4P, conv_b4,
                                           cw5P, conv_b5, pooled);
    mgmp_fc_semantic<<<16, 256, 0, stream>>>(pooled, fc_w, fc_b, semantic);
    // 3) meta-path GRU encoders
    mgmp_gru<<<320, 128, 0, stream>>>(news_emb, source_emb, news_ids, nsn_mid, nsn_end,
                                      nsn_gru_wih, nsn_gru_whh, nsn_gru_bih, nsn_gru_bhh,
                                      enc);
    mgmp_gru<<<320, 128, 0, stream>>>(news_emb, user_emb, news_ids, nun_mid, nun_end,
                                      nun_gru_wih, nun_gru_whh, nun_gru_bih, nun_gru_bhh,
                                      enc + 320 * 128);
    // 4) sample MHSA + LN + mean
    mgmp_samp_attn<<<32, 128, 0, stream>>>(enc, nsn_aw, nsn_ab, nsn_ow, nsn_ob,
                                           nsn_lg, nsn_lb, evec);
    mgmp_samp_attn<<<32, 128, 0, stream>>>(enc + 320 * 128, nun_aw, nun_ab, nun_ow, nun_ob,
                                           nun_lg, nun_lb, evec + 32 * 128);
    // 5) meta-path fusion + classifier
    mgmp_final<<<32, 128, 0, stream>>>(semantic, evec, evec + 32 * 128,
                                       mp_w1, mp_b1, mp_w2, mp_b2,
                                       cls_w1, cls_b1, cls_w2, cls_b2,
                                       (float*)d_out);
}
